// SE3CGNonLinearity_10282151706924
// MI455X (gfx1250) — matrix operand order repaired and checked by reading, not compiled, not run on hardware
//
#include <hip/hip_runtime.h>

#define C_CH 32

typedef __attribute__((ext_vector_type(16))) __bf16  v16bf;
typedef __attribute__((ext_vector_type(8)))  float   v8f;

// ---------------- CG pair tables (l output, l1, l2, flat CG offset) ----------
// l=0: (0,0),(1,1),(2,2)   l=1: (0,1),(1,0),(1,1),(1,2),(2,1),(2,2)
// l=2: (0,2),(1,1),(1,2),(2,0),(2,1),(2,2)   (reference iteration order)
#define NPAIR_TOT 15
__device__ const int PAIR_L [NPAIR_TOT] = {0,0,0, 1,1,1,1,1,1, 2,2,2,2,2,2};
__device__ const int PAIR_L1[NPAIR_TOT] = {0,1,2, 0,1,1,1,2,2, 0,1,1,2,2,2};
__device__ const int PAIR_L2[NPAIR_TOT] = {0,1,2, 1,0,1,2,1,2, 2,1,2,0,1,2};
__device__ const int PAIR_CGO[NPAIR_TOT]= {0,1,10, 35,44,53,80,125,170, 245,270,315,390,415,490};
#define CG_TOTAL 615
// LDS base (floats) of the re-part region for each l; im region = re + 512*(2l+1)
__device__ const int XRE[3] = {0, 1024, 4096};

__device__ float g_cg[CG_TOTAL];
// Pre-swizzled bf16 weights in WMMA B-fragment layout:
// 30 tiles (pp*2+nt), each tile = re(1024B) + im(1024B); within a 1024B part:
// lane L's low 16B at +L*16, high 16B at +512+L*16  -> fully coalesced b128 loads.
__device__ __align__(16) unsigned char g_wpack[NPAIR_TOT * 2 * 2048];

// ---------------- exact Clebsch-Gordan coefficients, computed on device ------
__device__ double cg_coef(int j1, int m1, int j2, int m2, int j, int m) {
  if (m1 + m2 != m) return 0.0;
  double f[8];
  f[0] = 1.0;
  for (int i = 1; i < 8; ++i) f[i] = f[i-1] * (double)i;
  double pref = sqrt((2.0*j + 1.0) * f[j+j1-j2] * f[j-j1+j2] * f[j1+j2-j] / f[j1+j2+j+1]);
  pref *= sqrt(f[j+m] * f[j-m] * f[j1-m1] * f[j1+m1] * f[j2-m2] * f[j2+m2]);
  double s = 0.0;
  for (int k = 0; k <= j1 + j2 - j; ++k) {
    int d0 = k, d1 = j1+j2-j-k, d2 = j1-m1-k, d3 = j2+m2-k, d4 = j-j2+m1+k, d5 = j-j1-m2+k;
    if (d0 < 0 || d1 < 0 || d2 < 0 || d3 < 0 || d4 < 0 || d5 < 0) continue;
    double term = f[d0]*f[d1]*f[d2]*f[d3]*f[d4]*f[d5];
    s += ((k & 1) ? -1.0 : 1.0) / term;
  }
  return pref * s;
}

__global__ void cg_init_kernel() {
  if (threadIdx.x != 0 || blockIdx.x != 0) return;
  for (int i = 0; i < CG_TOTAL; ++i) g_cg[i] = 0.0f;
  for (int pp = 0; pp < NPAIR_TOT; ++pp) {
    const int l = PAIR_L[pp], l1 = PAIR_L1[pp], l2 = PAIR_L2[pp];
    const int d2 = 2*l2 + 1, dl = 2*l + 1;
    for (int m1 = -l1; m1 <= l1; ++m1)
      for (int m2 = -l2; m2 <= l2; ++m2) {
        const int m = m1 + m2;
        if (m < -l || m > l) continue;
        g_cg[PAIR_CGO[pp] + ((m1+l1)*d2 + (m2+l2))*dl + (m+l)] =
            (float)cg_coef(l1, m1, l2, m2, l, m);
      }
  }
}

// ---------------- helpers ----------------------------------------------------
__device__ __forceinline__ unsigned short bfbits(float x) {   // hw cvt f32->bf16
  __bf16 h = (__bf16)x;
  return __builtin_bit_cast(unsigned short, h);
}

union BfVec {
  uint4          q[2];
  unsigned int   w[8];
  unsigned short u[16];
  __bf16         h[16];
  v16bf          v;
};

// ---------------- one-time weight pack: fp32 W -> bf16 B-fragments -----------
__global__ __launch_bounds__(32) void wpack_kernel(
    const float* __restrict__ w0re, const float* __restrict__ w0im,
    const float* __restrict__ w1re, const float* __restrict__ w1im,
    const float* __restrict__ w2re, const float* __restrict__ w2im) {
  const int tid = blockIdx.x;                 // 0..29 = pp*2 + nt
  const int pp = tid >> 1, nt = tid & 1;
  const int l = PAIR_L[pp];
  const int ploc = pp - ((l == 0) ? 0 : ((l == 1) ? 3 : 9));
  const float* wre = (l == 0) ? w0re : ((l == 1) ? w1re : w2re);
  const float* wim = (l == 0) ? w0im : ((l == 1) ? w1im : w2im);
  const int lane = threadIdx.x;
  const int half = lane >> 4, r = lane & 15;

  BfVec pre, pim;
#pragma unroll
  for (int s = 0; s < 16; ++s) {
    const int kk  = ((s < 8) ? s : s + 8) + half*8;  // lanes 0-15: K {0..7,16..23}; 16-31: +8
    const int row = ploc*C_CH + kk;
    const int col = nt*16 + r;
    pre.u[s] = bfbits(wre[row*C_CH + col]);
    pim.u[s] = bfbits(wim[row*C_CH + col]);
  }
  uint4* dre = (uint4*)(g_wpack + (size_t)tid * 2048);
  uint4* dim_ = (uint4*)(g_wpack + (size_t)tid * 2048 + 1024);
  dre[lane] = pre.q[0];  dre[32 + lane] = pre.q[1];
  dim_[lane] = pim.q[0]; dim_[32 + lane] = pim.q[1];
}

// ---------------- per-l complex GEMM through bf16 WMMA -----------------------
template <int L>
__device__ __forceinline__ void compute_l(const float* __restrict__ sX,
                                          const float* __restrict__ sCG,
                                          float2* __restrict__ out2,
                                          int base_pt, int lane) {
  constexpr int ROWS = 2*L + 1;               // A rows per point, also #M tiles
  constexpr int NP   = (L == 0) ? 3 : 6;      // CG pairs = K blocks of 32
  constexpr int PST  = (L == 0) ? 0 : ((L == 1) ? 3 : 9);
  constexpr int QOFF = (L == 0) ? 0 : ((L == 1) ? 32 : 128);
  const int half = lane >> 4;
  const int r    = lane & 15;

  for (int mt = 0; mt < ROWS; ++mt) {         // M tile of 16 (pt,m) rows
    v8f accRe[2] = {};
    v8f accIm[2] = {};
    const int g   = mt*16 + r;                // this lane's A row
    const int ptl = g / ROWS;
    const int m   = g % ROWS;

#pragma unroll
    for (int pl = 0; pl < NP; ++pl) {         // K blocks (one per CG pair)
      const int l1  = PAIR_L1[PST + pl];
      const int l2  = PAIR_L2[PST + pl];
      const int d1  = 2*l1 + 1, d2 = 2*l2 + 1;
      const int cgo = PAIR_CGO[PST + pl];
      const int x1re0 = XRE[l1];
      const int x2re0 = XRE[l2];

      // ---- build complex A fragment (16x32 bf16 tile, K slot = channel) ----
      float fr[16] = {};
      float fi[16] = {};
#pragma unroll 1
      for (int a = 0; a < d1; ++a) {
        const int b1  = x1re0 + (ptl*d1 + a)*C_CH;
        const int b1i = b1 + 512*d1;
#pragma unroll 1
        for (int b = 0; b < d2; ++b) {
          const float t  = sCG[cgo + (a*d2 + b)*ROWS + m];
          const int  b2  = x2re0 + (ptl*d2 + b)*C_CH;
          const int  b2i = b2 + 512*d2;
#pragma unroll
          for (int s = 0; s < 16; ++s) {
            // A layout: lanes 0-15 hold K {0..7,16..23}; lanes 16-31 +8
            const int c = ((s < 8) ? s : s + 8) + half*8;
            const float ar = sX[b1  + c], ai = sX[b1i + c];
            const float br = sX[b2  + c], bi = sX[b2i + c];
            fr[s] = fmaf(t, ar*br - ai*bi, fr[s]);
            fi[s] = fmaf(t, ar*bi + ai*br, fi[s]);
          }
        }
      }
      BfVec Are, Aim;
#pragma unroll
      for (int s = 0; s < 16; ++s) { Are.h[s] = (__bf16)fr[s]; Aim.h[s] = (__bf16)fi[s]; }

      // ---- load pre-swizzled complex B fragments (coalesced b128) ----------
#pragma unroll
      for (int nt = 0; nt < 2; ++nt) {
        const uint4* tre = (const uint4*)(g_wpack + (size_t)((PST + pl)*2 + nt)*2048);
        const uint4* tim = tre + 64;          // +1024 bytes
        BfVec Bre, Bim, BimN;
        Bre.q[0] = tre[lane];  Bre.q[1] = tre[32 + lane];
        Bim.q[0] = tim[lane];  Bim.q[1] = tim[32 + lane];
#pragma unroll
        for (int i = 0; i < 8; ++i) BimN.w[i] = Bim.w[i] ^ 0x80008000u;  // -Bim

        // ---- complex MMA: 4 real WMMAs, fp32 accumulate --------------------
        accRe[nt] = __builtin_amdgcn_wmma_f32_16x16x32_bf16(false, Are.v, false, Bre.v,
                                                            (short)0, accRe[nt], false, false);
        accRe[nt] = __builtin_amdgcn_wmma_f32_16x16x32_bf16(false, Aim.v, false, BimN.v,
                                                            (short)0, accRe[nt], false, false);
        accIm[nt] = __builtin_amdgcn_wmma_f32_16x16x32_bf16(false, Are.v, false, Bim.v,
                                                            (short)0, accIm[nt], false, false);
        accIm[nt] = __builtin_amdgcn_wmma_f32_16x16x32_bf16(false, Aim.v, false, Bre.v,
                                                            (short)0, accIm[nt], false, false);
      }
    }

    // ---- store: D layout VGPR v: (M = v + 8*half, N = r) ------------------
#pragma unroll
    for (int nt = 0; nt < 2; ++nt) {
#pragma unroll
      for (int vg = 0; vg < 8; ++vg) {
        const int Mrow = vg + 8*half;
        const int gg = mt*16 + Mrow;
        const int pt = gg / ROWS, mm = gg % ROWS;
        const int q  = QOFF + mm*C_CH + nt*16 + r;
        out2[(size_t)(base_pt + pt)*288 + q] = make_float2(accRe[nt][vg], accIm[nt][vg]);
      }
    }
  }
}

// ---------------- main kernel: 1 wave32 = 16 points --------------------------
__global__ __launch_bounds__(32) void se3_cg_kernel(
    const float* __restrict__ x0re, const float* __restrict__ x0im,
    const float* __restrict__ x1re, const float* __restrict__ x1im,
    const float* __restrict__ x2re, const float* __restrict__ x2im,
    float* __restrict__ out, int npt) {
  __shared__ __align__(16) float sX[9216];    // 16 pts x 9 m x 32 ch x {re,im}
  __shared__ float sCG[CG_TOTAL];
  const int lane    = threadIdx.x;
  const int base_pt = blockIdx.x * 16;
  if (base_pt >= npt) return;

  // speculative prefetch of next tile's largest stream -> global_prefetch_b8
  if (base_pt + 16 < npt) {
    __builtin_prefetch(x2re + (size_t)(base_pt + 16)*5*C_CH, 0, 0);
    __builtin_prefetch(x2im + (size_t)(base_pt + 16)*5*C_CH, 0, 0);
  }

  // stage x to LDS with CDNA5 async copies (GVS mode: sgpr base + vgpr offset).
  // LDS byte address = low 32 bits of the flat shared-pointer (aperture map).
  const float* xs[6] = {x0re, x0im, x1re, x1im, x2re, x2im};
  const int    xo[6] = {0, 512, 1024, 2560, 4096, 6656};
#pragma unroll
  for (int i = 0; i < 6; ++i) {
    const int l   = i >> 1;
    const int dim = 2*l + 1;
    const int n16 = dim * 128;                // # of 16-byte chunks in this stream
    const unsigned long long src =
        (unsigned long long)(xs[i] + (size_t)base_pt*dim*C_CH);
    const unsigned lds0 = (unsigned)(unsigned long long)(const void*)(sX + xo[i]);
    for (int t = lane; t < n16; t += 32) {
      const unsigned goff  = (unsigned)t * 16u;
      const unsigned laddr = lds0 + (unsigned)t * 16u;
      asm volatile("global_load_async_to_lds_b128 %0, %1, %2"
                   :: "v"(laddr), "v"(goff), "s"(src) : "memory");
    }
  }
  for (int t = lane; t < CG_TOTAL; t += 32) sCG[t] = g_cg[t];
  asm volatile("s_wait_asynccnt 0x0" ::: "memory");
  __syncthreads();

  float2* out2 = (float2*)out;
  compute_l<0>(sX, sCG, out2, base_pt, lane);
  compute_l<1>(sX, sCG, out2, base_pt, lane);
  compute_l<2>(sX, sCG, out2, base_pt, lane);
}

// ---------------- host entry -------------------------------------------------
extern "C" void kernel_launch(void* const* d_in, const int* in_sizes, int n_in,
                              void* d_out, int out_size, void* d_ws, size_t ws_size,
                              hipStream_t stream) {
  // setup_inputs() dict order: per l -> x{l}_re, x{l}_im, w{l}_re, w{l}_im
  const float* x0re = (const float*)d_in[0];
  const float* x0im = (const float*)d_in[1];
  const float* w0re = (const float*)d_in[2];
  const float* w0im = (const float*)d_in[3];
  const float* x1re = (const float*)d_in[4];
  const float* x1im = (const float*)d_in[5];
  const float* w1re = (const float*)d_in[6];
  const float* w1im = (const float*)d_in[7];
  const float* x2re = (const float*)d_in[8];
  const float* x2im = (const float*)d_in[9];
  const float* w2re = (const float*)d_in[10];
  const float* w2im = (const float*)d_in[11];

  const int npt = in_sizes[0] / C_CH;         // B*N (x0_re is [B,N,1,32])
  hipLaunchKernelGGL(cg_init_kernel, dim3(1), dim3(32), 0, stream);
  hipLaunchKernelGGL(wpack_kernel, dim3(NPAIR_TOT * 2), dim3(32), 0, stream,
                     w0re, w0im, w1re, w1im, w2re, w2im);
  const int blocks = (npt + 15) / 16;
  hipLaunchKernelGGL(se3_cg_kernel, dim3(blocks), dim3(32), 0, stream,
                     x0re, x0im, x1re, x1im, x2re, x2im, (float*)d_out, npt);
  (void)n_in; (void)out_size; (void)d_ws; (void)ws_size;
}
